// ContrastDrivenFeatureAlignment_47742856462452
// MI455X (gfx1250) — matrix operand order repaired and checked by reading, not compiled, not run on hardware
//
#include <hip/hip_runtime.h>

#define B_    4
#define C_    64
#define H_    128
#define W_    128
#define HW_   (H_*W_)        // 16384
#define CL_   (C_*HW_)       // 1048576 = 2^20
#define TELEM (B_*CL_)       // 4194304
#define SCALE_ 0.125f        // 64^-0.5

typedef __attribute__((ext_vector_type(16))) _Float16 v16h;
typedef __attribute__((ext_vector_type(8)))  _Float16 v8h;
typedef __attribute__((ext_vector_type(8)))  float    v8f;

// ---------------------------------------------------------------------------
// 1x1 conv (64->64) as WMMA GEMM.  Grid: 512 blocks x 256 threads.
// Block = 128 contiguous pixels of one batch; 8 waves, each wave owns a
// 16-pixel strip and computes all 64 output channels (4 WMMA row-tiles).
// The f32 weight tile is DMA'd raw into LDS with the CDNA5 async-to-LDS
// engine (ASYNCcnt), overlapped with the f32->f16 transpose staging of the
// input tile; WMMA operand fetches are ds_load_b128; accumulation in f32.
// ---------------------------------------------------------------------------
__global__ __launch_bounds__(256)
void k_conv1x1_wmma(const float* __restrict__ in, const float* __restrict__ w,
                    const float* __restrict__ bias, float* __restrict__ out)
{
  __shared__ __align__(16) float    sWraw[C_ * C_];  // async DMA dest, 16 KB
  __shared__ __align__(32) _Float16 sW[C_ * C_];     // [o][k]          8 KB
  __shared__ __align__(32) _Float16 sX[128 * C_];    // [pix][k]       16 KB
  __shared__ float sB[C_];

  const int t       = threadIdx.x;          // 0..255
  const int blk     = blockIdx.x;           // 0..511
  const int b       = blk >> 7;             // 128 blocks per batch
  const int pixbase = (blk & 127) * 128;
  const float* inb  = in + (size_t)b * C_ * HW_;

  // ---- 1) kick off async global->LDS DMA of the raw f32 weights ----------
  {
    unsigned lds0 = (unsigned)(uintptr_t)&sWraw[0];  // LDS aperture: low 32b
#pragma unroll
    for (int it = 0; it < 4; ++it) {
      unsigned byteoff = (unsigned)(it * 4096 + t * 16);
      unsigned ldsa = lds0 + byteoff;
      unsigned long long ga =
          (unsigned long long)(uintptr_t)w + (unsigned long long)byteoff;
      asm volatile("global_load_async_to_lds_b128 %0, %1, off"
                   :: "v"(ldsa), "v"(ga) : "memory");
    }
  }

  // ---- 2) overlap: 64ch x 128px input tile, float4 coalesced loads,
  //         f32->f16 transposed into sX[pix][k] ----------------------------
#pragma unroll
  for (int it = 0; it < 8; ++it) {
    int lin = it * 1024 + t * 4;
    int c   = lin >> 7;
    int i   = lin & 127;
    float4 v = *(const float4*)(inb + (size_t)c * HW_ + pixbase + i);
    sX[(i + 0) * C_ + c] = (_Float16)v.x;
    sX[(i + 1) * C_ + c] = (_Float16)v.y;
    sX[(i + 2) * C_ + c] = (_Float16)v.z;
    sX[(i + 3) * C_ + c] = (_Float16)v.w;
  }

  // ---- 3) wait for weight DMA, convert to f16 ----------------------------
  asm volatile("s_wait_asynccnt 0" ::: "memory");
  for (int i = t; i < C_ * C_; i += 256) sW[i] = (_Float16)sWraw[i];
  if (t < C_) sB[t] = bias[t];
  __syncthreads();

  const int lane = t & 31;
  const int wave = t >> 5;
  const int n    = lane & 15;      // A row (out-ch low bits), B/D col (pixel)
  const int half = lane >> 4;
  const int pl   = wave * 16 + n;  // local pixel for this lane's B column

  v8f acc[4] = {};

#pragma unroll
  for (int kb = 0; kb < C_; kb += 32) {
    // B 32x16: element e -> K = half*16 + e  (16 contiguous halfs, 32B)
    v16h bm = *(const v16h*)&sX[pl * C_ + kb + half * 16];
#pragma unroll
    for (int tt = 0; tt < 4; ++tt) {
      int o = tt * 16 + n;  // A row
      // A 16x32: element e -> K = half*16 + e (matches B)
      v8h lo = *(const v8h*)&sW[o * C_ + kb + half * 16];
      v8h hi = *(const v8h*)&sW[o * C_ + kb + half * 16 + 8];
      v16h am = __builtin_shufflevector(lo, hi, 0,1,2,3,4,5,6,7,
                                               8,9,10,11,12,13,14,15);
      acc[tt] = __builtin_amdgcn_wmma_f32_16x16x32_f16(
          false, am, false, bm, (short)0, acc[tt], false, false);
    }
  }

  // D 16x16 f32: VGPR r -> row M = r + half*8, col N = n
  float* outb = out + (size_t)b * C_ * HW_ + pixbase + wave * 16 + n;
#pragma unroll
  for (int tt = 0; tt < 4; ++tt)
#pragma unroll
    for (int r = 0; r < 8; ++r) {
      int o = tt * 16 + half * 8 + r;
      outb[(size_t)o * HW_] = acc[tt][r] + sB[o];
    }
}

// ---------------------------------------------------------------------------
// Attention (softmax over 9 consecutive flat elements) + permuted fold
// scatter-add.  One thread per softmax group (b, m), m in [0, C*L).
// Within a group: channel c = m>>14 is constant; element p' has
// idx = 9*(m&16383)+p', p = idx>>14, l = idx&16383 (the unfold gather).
// Fold target for p': r = p'*64 + c, cf = r/9, wp = r%9, s = m&16383.
// ---------------------------------------------------------------------------
__global__ __launch_bounds__(256)
void k_attn_fold(const float* __restrict__ xq, const float* __restrict__ fgp,
                 const float* __restrict__ bgp,
                 float* __restrict__ of, float* __restrict__ ob)
{
  unsigned gid = blockIdx.x * 256u + threadIdx.x;   // < TELEM
  unsigned m   = gid & (CL_ - 1u);
  unsigned b   = gid >> 20;
  unsigned c   = m >> 14;           // channel (== q in the permuted reshape)
  unsigned l0  = m & 16383u;        // spatial col s of the fold input

  const size_t basebc = ((size_t)b * C_ + c) * HW_;
  const float* xp = xq  + basebc;
  const float* fp = fgp + basebc;
  const float* gp = bgp + basebc;

  float xv[9], sf[9], sb[9];
  unsigned rem = 9u * l0;
#pragma unroll
  for (int pp = 0; pp < 9; ++pp) {
    unsigned idx = rem + pp;
    unsigned p   = idx >> 14;          // window position of unfold
    unsigned l   = idx & 16383u;       // spatial flat
    int hy = (int)(l >> 7)   + (int)(p / 3u) - 1;
    int wx = (int)(l & 127u) + (int)(p % 3u) - 1;
    float x_ = 0.f, f_ = 0.f, g_ = 0.f;
    if ((unsigned)hy < H_ && (unsigned)wx < W_) {
      int off = hy * W_ + wx;
      x_ = xp[off]; f_ = fp[off]; g_ = gp[off];
    }
    xv[pp] = x_;
    sf[pp] = x_ * f_;
    sb[pp] = x_ - g_;
  }

  float mf = sf[0], mb = sb[0];
#pragma unroll
  for (int i = 1; i < 9; ++i) { mf = fmaxf(mf, sf[i]); mb = fmaxf(mb, sb[i]); }
  float df = 0.f, db = 0.f;
#pragma unroll
  for (int i = 0; i < 9; ++i) {
    sf[i] = __expf(sf[i] - mf); df += sf[i];
    sb[i] = __expf(sb[i] - mb); db += sb[i];
  }
  float rf = 1.f / df, rb = 1.f / db;

  unsigned hs = l0 >> 7, ws = l0 & 127u;
#pragma unroll
  for (int pp = 0; pp < 9; ++pp) {
    float af = sf[pp] * rf * xv[pp] * SCALE_;
    float ab = sb[pp] * rb * xv[pp] * SCALE_;
    unsigned r  = (unsigned)pp * 64u + c;
    unsigned cf = r / 9u;
    unsigned wp = r - cf * 9u;
    int oy = (int)(wp / 3u) + (int)hs - 1;
    int ox = (int)(wp % 3u) + (int)ws - 1;
    if ((unsigned)oy < H_ && (unsigned)ox < W_) {
      size_t o = ((size_t)b * C_ + cf) * HW_ + (unsigned)oy * W_ + (unsigned)ox;
      atomicAdd(of + o, af);
      atomicAdd(ob + o, ab);
    }
  }
}

// ---------------------------------------------------------------------------
// grouped 3x3 conv over concat(of,ob) (groups=64, 2 in-ch per group:
// concat channels 2o and 2o+1) + bias + eval-BN + ReLU.
// ---------------------------------------------------------------------------
__global__ __launch_bounds__(256)
void k_gconv2_bn_relu(const float* __restrict__ of, const float* __restrict__ ob,
                      const float* __restrict__ w1, const float* __restrict__ b1,
                      const float* __restrict__ g1, const float* __restrict__ be1,
                      float* __restrict__ y)
{
  unsigned gid = blockIdx.x * 256u + threadIdx.x;   // NCHW flat
  unsigned b = gid >> 20;
  unsigned o = (gid >> 14) & 63u;
  unsigned l = gid & 16383u;
  int yy = (int)(l >> 7), xx = (int)(l & 127u);

  float sum = 0.f;
#pragma unroll
  for (int ci = 0; ci < 2; ++ci) {
    unsigned ch = 2u * o + (unsigned)ci;
    const float* src = (ch < 64u) ? of + ((size_t)b * C_ + ch) * HW_
                                  : ob + ((size_t)b * C_ + (ch - 64u)) * HW_;
    const float* wp = w1 + (o * 2u + (unsigned)ci) * 9u;
#pragma unroll
    for (int i = 0; i < 3; ++i)
#pragma unroll
      for (int j = 0; j < 3; ++j) {
        int py = yy + i - 1, px = xx + j - 1;
        if ((unsigned)py < H_ && (unsigned)px < W_)
          sum += wp[i * 3 + j] * src[py * W_ + px];
      }
  }
  float s = g1[o] * rsqrtf(1.f + 1e-5f);
  float v = (sum + b1[o]) * s + be1[o];
  y[gid] = fmaxf(v, 0.f);
}

// ---------------------------------------------------------------------------
// depthwise 3x3 conv + bias + eval-BN + ReLU
// ---------------------------------------------------------------------------
__global__ __launch_bounds__(256)
void k_dwconv_bn_relu(const float* __restrict__ in, const float* __restrict__ w3,
                      const float* __restrict__ b3, const float* __restrict__ g2,
                      const float* __restrict__ be2, float* __restrict__ y)
{
  unsigned gid = blockIdx.x * 256u + threadIdx.x;
  unsigned b = gid >> 20;
  unsigned o = (gid >> 14) & 63u;
  unsigned l = gid & 16383u;
  int yy = (int)(l >> 7), xx = (int)(l & 127u);

  const float* src = in + ((size_t)b * C_ + o) * HW_;
  const float* wp  = w3 + o * 9u;
  float sum = 0.f;
#pragma unroll
  for (int i = 0; i < 3; ++i)
#pragma unroll
    for (int j = 0; j < 3; ++j) {
      int py = yy + i - 1, px = xx + j - 1;
      if ((unsigned)py < H_ && (unsigned)px < W_)
        sum += wp[i * 3 + j] * src[py * W_ + px];
    }
  float s = g2[o] * rsqrtf(1.f + 1e-5f);
  float v = (sum + b3[o]) * s + be2[o];
  y[gid] = fmaxf(v, 0.f);
}

// ---------------------------------------------------------------------------
extern "C" void kernel_launch(void* const* d_in, const int* in_sizes, int n_in,
                              void* d_out, int out_size, void* d_ws, size_t ws_size,
                              hipStream_t stream) {
  (void)in_sizes; (void)n_in; (void)out_size; (void)ws_size;

  const float* x      = (const float*)d_in[0];
  const float* fg     = (const float*)d_in[1];
  const float* bg     = (const float*)d_in[2];
  const float* w_x    = (const float*)d_in[3];
  const float* bias_x = (const float*)d_in[4];
  const float* w_f    = (const float*)d_in[5];
  const float* bias_f = (const float*)d_in[6];
  const float* w_b    = (const float*)d_in[7];
  const float* bias_b = (const float*)d_in[8];
  const float* w1     = (const float*)d_in[9];
  const float* b1     = (const float*)d_in[10];
  const float* g1     = (const float*)d_in[11];
  const float* be1    = (const float*)d_in[12];
  const float* w2     = (const float*)d_in[13];
  const float* b2     = (const float*)d_in[14];
  const float* w3     = (const float*)d_in[15];
  const float* b3     = (const float*)d_in[16];
  const float* g2     = (const float*)d_in[17];
  const float* be2    = (const float*)d_in[18];
  const float* w4     = (const float*)d_in[19];
  const float* b4     = (const float*)d_in[20];
  float* out = (float*)d_out;

  float* xq  = (float*)d_ws;          // 16 MB each
  float* fgp = xq  + TELEM;
  float* bgp = fgp + TELEM;
  float* of  = bgp + TELEM;
  float* ob  = of  + TELEM;
  float* y1  = xq;    // reuse after attention pass
  float* y2  = fgp;
  float* y3  = bgp;

  dim3 blk(256);
  const int gemmGrid = 512;            // (B*HW/128) blocks
  const int eltGrid  = TELEM / 256;    // 16384 blocks

  k_conv1x1_wmma<<<gemmGrid, blk, 0, stream>>>(x,  w_x, bias_x, xq);
  k_conv1x1_wmma<<<gemmGrid, blk, 0, stream>>>(fg, w_f, bias_f, fgp);
  k_conv1x1_wmma<<<gemmGrid, blk, 0, stream>>>(bg, w_b, bias_b, bgp);

  hipMemsetAsync(of, 0, 2 * (size_t)TELEM * sizeof(float), stream);

  k_attn_fold<<<eltGrid, blk, 0, stream>>>(xq, fgp, bgp, of, ob);
  k_gconv2_bn_relu<<<eltGrid, blk, 0, stream>>>(of, ob, w1, b1, g1, be1, y1);
  k_conv1x1_wmma<<<gemmGrid, blk, 0, stream>>>(y1, w2, b2, y2);
  k_dwconv_bn_relu<<<eltGrid, blk, 0, stream>>>(y2, w3, b3, g2, be2, y3);
  k_conv1x1_wmma<<<gemmGrid, blk, 0, stream>>>(y3, w4, b4, out);
}
